// CnnAttnLstmCRF_33947421508249
// MI455X (gfx1250) — compile-verified, matrix operand order checked by reading, not run on hardware
//
#include <hip/hip_runtime.h>

// ---------------------------------------------------------------------------
// CnnAttnLstmCRF on gfx1250 (MI455X): all dense GEMMs via v_wmma_f32_16x16x32_f16
// (f16 operands, f32 accumulate), sequential scans as LDS-resident kernels.
// GEMM waves compute 16x64 tiles: A fragment reused across 4 WMMAs per k-chunk.
// ---------------------------------------------------------------------------

static constexpr int kB = 64;
static constexpr int kT = 256;
static constexpr int kG = 66;
static constexpr int kSTART = 64;
static constexpr int kSTOP  = 65;
static constexpr int kDC = 128, kDW = 256, kDL = 128, kDI = 128;
static constexpr int kHC = 384;
static constexpr int kLH = 512;     // bi-LSTM output width
static constexpr int kH  = 256;     // per-direction hidden
static constexpr int kH4 = 1024;    // 4*H gates
static constexpr int kD  = 512;     // q/kv feature width
static constexpr int kGP = 80;      // G padded to multiple of 16 (emit width)
static constexpr int kGPad = 128;   // proj_w rows padded to multiple of 64
static constexpr int kBT = kB * kT;

typedef __attribute__((ext_vector_type(16))) _Float16 v16h;
typedef __attribute__((ext_vector_type(8)))  float    v8f;
typedef __attribute__((ext_vector_type(8)))  unsigned int v8u;

// ---------------------------------------------------------------------------
// Generic batched NT GEMM: C[M,N] = A[M,K] * B[N,K]^T (+bias[n]), via WMMA.
// One wave computes a 16x64 tile (4 accumulators, A fragment reused 4x).
// K must be a multiple of 32; M a multiple of 16; B must be readable up to
// the 64-row tile boundary (callers pad); stores guarded by n < N.
// ---------------------------------------------------------------------------
__global__ void gemm_nt_wmma(const _Float16* __restrict__ A,
                             const _Float16* __restrict__ B,
                             float* __restrict__ Cf, _Float16* __restrict__ Ch,
                             int M, int N, int K,
                             long long sA, long long sB, long long sC,
                             const float* __restrict__ bias, int biasLen) {
  const int lane = threadIdx.x & 31;
  const int tn = blockIdx.x;   // 64-wide N tile
  const int tm = blockIdx.y;   // 16-wide M tile
  const int bz = blockIdx.z;

  const _Float16* Ab = A + (size_t)bz * sA;
  const _Float16* Bb = B + (size_t)bz * sB;

  const int rlo = lane & 15;
  const int h8  = ((lane >> 4) & 1) * 8;
  const _Float16* arow  = Ab + (size_t)(tm * 16 + rlo) * K;
  const _Float16* brow0 = Bb + (size_t)(tn * 64 + rlo) * K;

  v8f acc[4] = {};
  for (int k0 = 0; k0 < K; k0 += 32) {
    v8u ua;
#pragma unroll
    for (int v = 0; v < 8; ++v) {
      // ISA 16-bit A 16x32 layout: VGPR v holds K = (v&3)*2 + (lane>=16?8:0) + (v>=4?16:0)
      const int k = k0 + ((v & 3) * 2) + h8 + ((v >> 2) * 16);
      ua[v] = *(const unsigned int*)(arow + k);
    }
    const v16h a = __builtin_bit_cast(v16h, ua);
#pragma unroll
    for (int s = 0; s < 4; ++s) {
      v8u ub;
#pragma unroll
      for (int v = 0; v < 8; ++v) {
        const int k = k0 + ((v & 3) * 2) + h8 + ((v >> 2) * 16);
        ub[v] = *(const unsigned int*)(brow0 + (size_t)s * 16 * K + k);
      }
      const v16h b = __builtin_bit_cast(v16h, ub);
      acc[s] = __builtin_amdgcn_wmma_f32_16x16x32_f16(false, a, false, b,
                                                      (short)0, acc[s], false, false);
    }
  }

  // D layout: n = lane&15 ; m = r + (lane>=16 ? 8 : 0)
  const int mbase = tm * 16 + h8;
#pragma unroll
  for (int s = 0; s < 4; ++s) {
    const int n = tn * 64 + s * 16 + rlo;
#pragma unroll
    for (int r = 0; r < 8; ++r) {
      const int m = mbase + r;
      if (m < M && n < N) {
        float val = acc[s][r];
        if (bias && n < biasLen) val += bias[n];
        const size_t idx = (size_t)bz * sC + (size_t)m * N + n;
        if (Cf) Cf[idx] = val;
        else    Ch[idx] = (_Float16)val;
      }
    }
  }
}

// ---------------------------------------------------------------------------
// Small prep kernels
// ---------------------------------------------------------------------------
__global__ void cvt_f32_to_f16(const float* __restrict__ src,
                               _Float16* __restrict__ dst, int n) {
  for (int i = blockIdx.x * blockDim.x + threadIdx.x; i < n;
       i += gridDim.x * blockDim.x)
    dst[i] = (_Float16)src[i];
}

__global__ void pad_proj_w(const float* __restrict__ pw, _Float16* __restrict__ dst) {
  int i = blockIdx.x * blockDim.x + threadIdx.x;   // over 128*512
  if (i < kGPad * kLH) {
    int o = i / kLH, d = i % kLH;
    dst[i] = (o < kG) ? (_Float16)pw[o * kLH + d] : (_Float16)0.0f;
  }
}

__global__ void comb_bias(const float* __restrict__ bih,
                          const float* __restrict__ bhh,
                          float* __restrict__ nb, int n) {
  int i = blockIdx.x * blockDim.x + threadIdx.x;
  if (i < n) nb[i] = bih[i] + bhh[i];
}

// kv = [word(256) | lexi(128) | intent(128)]; also build kv^T (per-batch D x T)
__global__ void prep_kv(const int* __restrict__ word, const int* __restrict__ lexi,
                        const int* __restrict__ intents,
                        const float* __restrict__ wemb, const float* __restrict__ lemb,
                        const float* __restrict__ iemb,
                        _Float16* __restrict__ kv, _Float16* __restrict__ kvT) {
  const int bt = blockIdx.x;
  const int b = bt / kT, t = bt % kT;
  const int w = word[bt], x = lexi[bt], ii = intents[b];
  for (int d = threadIdx.x; d < kD; d += blockDim.x) {
    float val;
    if (d < 256)      val = wemb[(size_t)w * kDW + d];
    else if (d < 384) val = lemb[(size_t)x * kDL + (d - 256)];
    else              val = iemb[(size_t)ii * kDI + (d - 384)];
    _Float16 hv = (_Float16)val;
    kv[(size_t)bt * kD + d] = hv;
    kvT[((size_t)b * kD + d) * kT + t] = hv;
  }
}

// im2col for k=3 "same" conv; entry e = c*3 + k matches conv_w (HC,DC,3) rows
__global__ void im2col_char(const int* __restrict__ bchar,
                            const float* __restrict__ cemb,
                            _Float16* __restrict__ A) {
  const int bt = blockIdx.x;
  const int b = bt / kT, t = bt % kT;
  for (int e = threadIdx.x; e < 3 * kDC; e += blockDim.x) {
    const int c = e / 3, k = e % 3;
    const int t2 = t + k - 1;
    float v = 0.0f;
    if (t2 >= 0 && t2 < kT) {
      const int ch = bchar[b * kT + t2];
      v = cemb[(size_t)ch * kDC + c];
    }
    A[(size_t)bt * (3 * kDC) + e] = (_Float16)v;
  }
}

// q = [char_cnn(384) | intent(128)]
__global__ void build_q(const float* __restrict__ convout,
                        const int* __restrict__ intents,
                        const float* __restrict__ iemb,
                        _Float16* __restrict__ q) {
  const int bt = blockIdx.x;
  const int b = bt / kT;
  const int ii = intents[b];
  for (int d = threadIdx.x; d < kD; d += blockDim.x) {
    float v = (d < kHC) ? convout[(size_t)bt * kHC + d]
                        : iemb[(size_t)ii * kDI + (d - kHC)];
    q[(size_t)bt * kD + d] = (_Float16)v;
  }
}

// softmax over k (row length 256) with 1/TEMP scaling; f16 output for WMMA
__global__ void softmax_rows(const float* __restrict__ S,
                             _Float16* __restrict__ P, float invTemp) {
  const int row = blockIdx.x;
  const int tid = threadIdx.x;
  __shared__ float red[256];
  const float v = S[(size_t)row * kT + tid] * invTemp;
  red[tid] = v;
  __syncthreads();
  for (int st = 128; st > 0; st >>= 1) {
    if (tid < st) red[tid] = fmaxf(red[tid], red[tid + st]);
    __syncthreads();
  }
  const float mx = red[0];
  __syncthreads();
  const float e = expf(v - mx);
  red[tid] = e;
  __syncthreads();
  for (int st = 128; st > 0; st >>= 1) {
    if (tid < st) red[tid] += red[tid + st];
    __syncthreads();
  }
  P[(size_t)row * kT + tid] = (_Float16)(e / red[0]);
}

// ---------------------------------------------------------------------------
// LSTM recurrence: one workgroup per batch element, h in LDS, 4 gates/thread.
// pre = x @ w_ih^T + (b_ih+b_hh) already computed; this adds h @ w_hh^T.
// Output h written as f16 into column slice of the next layer's x.
// ---------------------------------------------------------------------------
__global__ void lstm_recur(const float* __restrict__ pre,       // [B*T, 1024]
                           const float* __restrict__ whh,       // [1024, 256]
                           _Float16* __restrict__ xnext,        // [B*T, 512]
                           int reverse, int colOffset) {
  const int b = blockIdx.x;
  const int j = threadIdx.x;   // hidden unit 0..255
  __shared__ float h[kH];
  float c = 0.0f;
  h[j] = 0.0f;
  __syncthreads();

  const float* w_i = whh + (size_t)(0 * kH + j) * kH;
  const float* w_f = whh + (size_t)(1 * kH + j) * kH;
  const float* w_g = whh + (size_t)(2 * kH + j) * kH;
  const float* w_o = whh + (size_t)(3 * kH + j) * kH;

  for (int step = 0; step < kT; ++step) {
    const int t = reverse ? (kT - 1 - step) : step;
    const float* p = pre + ((size_t)b * kT + t) * kH4;
    float gi = p[j], gf = p[kH + j], gg = p[2 * kH + j], go = p[3 * kH + j];
    for (int k = 0; k < kH; ++k) {
      const float hk = h[k];
      gi += w_i[k] * hk;
      gf += w_f[k] * hk;
      gg += w_g[k] * hk;
      go += w_o[k] * hk;
    }
    const float si = 1.0f / (1.0f + expf(-gi));
    const float sf = 1.0f / (1.0f + expf(-gf));
    const float so = 1.0f / (1.0f + expf(-go));
    c = sf * c + si * tanhf(gg);
    const float hn = so * tanhf(c);
    __syncthreads();
    h[j] = hn;
    __syncthreads();
    xnext[((size_t)b * kT + t) * kLH + colOffset + j] = (_Float16)hn;
  }
}

// ---------------------------------------------------------------------------
// CRF forward (logZ) + gold score -> per-batch NLL
// ---------------------------------------------------------------------------
__global__ void crf_nll(const float* __restrict__ emit,    // [B*T, 80]
                        const float* __restrict__ trans,   // [66,66]
                        const int* __restrict__ labels,    // [B,T]
                        float* __restrict__ lossb) {
  const int b = blockIdx.x, tid = threadIdx.x;
  __shared__ float tr[kG * kG];
  __shared__ float alpha[kG], alpha2[kG];
  for (int i = tid; i < kG * kG; i += blockDim.x) tr[i] = trans[i];
  __syncthreads();
  if (tid < kG)
    alpha[tid] = tr[kSTART * kG + tid] + emit[((size_t)b * kT) * kGP + tid];
  __syncthreads();
  for (int t = 1; t < kT; ++t) {
    if (tid < kG) {
      float mx = -3.0e38f;
      for (int i = 0; i < kG; ++i) mx = fmaxf(mx, alpha[i] + tr[i * kG + tid]);
      float s = 0.0f;
      for (int i = 0; i < kG; ++i) s += expf(alpha[i] + tr[i * kG + tid] - mx);
      alpha2[tid] = mx + logf(s) + emit[((size_t)b * kT + t) * kGP + tid];
    }
    __syncthreads();
    if (tid < kG) alpha[tid] = alpha2[tid];
    __syncthreads();
  }
  if (tid == 0) {
    float mx = -3.0e38f;
    for (int g = 0; g < kG; ++g) mx = fmaxf(mx, alpha[g] + tr[g * kG + kSTOP]);
    float s = 0.0f;
    for (int g = 0; g < kG; ++g) s += expf(alpha[g] + tr[g * kG + kSTOP] - mx);
    const float logZ = mx + logf(s);
    const int* tg = labels + b * kT;
    float gold = tr[kSTART * kG + tg[0]];
    for (int t = 0; t < kT; ++t) gold += emit[((size_t)b * kT + t) * kGP + tg[t]];
    for (int t = 1; t < kT; ++t) gold += tr[tg[t - 1] * kG + tg[t]];
    gold += tr[tg[kT - 1] * kG + kSTOP];
    lossb[b] = logZ - gold;
  }
}

__global__ void reduce_loss(const float* __restrict__ lossb, float* __restrict__ out) {
  __shared__ float red[kB];
  red[threadIdx.x] = lossb[threadIdx.x];
  __syncthreads();
  for (int st = kB / 2; st > 0; st >>= 1) {
    if ((int)threadIdx.x < st) red[threadIdx.x] += red[threadIdx.x + st];
    __syncthreads();
  }
  if (threadIdx.x == 0) out[0] = red[0];
}

// ---------------------------------------------------------------------------
// Viterbi: forward pass with backpointers + thread-0 backtrace.
// Tags written as float (harness output dtype).
// ---------------------------------------------------------------------------
__global__ void viterbi(const float* __restrict__ emit,
                        const float* __restrict__ trans,
                        int* __restrict__ bp,           // [B, T-1, 66]
                        float* __restrict__ outTags) {  // [B, T]
  const int b = blockIdx.x, tid = threadIdx.x;
  __shared__ float tr[kG * kG];
  __shared__ float del[kG], del2[kG];
  for (int i = tid; i < kG * kG; i += blockDim.x) tr[i] = trans[i];
  __syncthreads();
  if (tid < kG)
    del[tid] = tr[kSTART * kG + tid] + emit[((size_t)b * kT) * kGP + tid];
  __syncthreads();
  for (int t = 1; t < kT; ++t) {
    if (tid < kG) {
      float best = -3.0e38f;
      int bi = 0;
      for (int i = 0; i < kG; ++i) {
        const float s = del[i] + tr[i * kG + tid];
        if (s > best) { best = s; bi = i; }
      }
      bp[((size_t)b * (kT - 1) + (t - 1)) * kG + tid] = bi;
      del2[tid] = best + emit[((size_t)b * kT + t) * kGP + tid];
    }
    __syncthreads();
    if (tid < kG) del[tid] = del2[tid];
    __syncthreads();
  }
  if (tid == 0) {
    float best = -3.0e38f;
    int last = 0;
    for (int g = 0; g < kG; ++g) {
      const float s = del[g] + tr[g * kG + kSTOP];
      if (s > best) { best = s; last = g; }
    }
    int tag = last;
    outTags[(size_t)b * kT + (kT - 1)] = (float)tag;
    for (int t = kT - 2; t >= 0; --t) {
      tag = bp[((size_t)b * (kT - 1) + t) * kG + tag];
      outTags[(size_t)b * kT + t] = (float)tag;
    }
  }
}

// ---------------------------------------------------------------------------
// Host launcher
// ---------------------------------------------------------------------------
extern "C" void kernel_launch(void* const* d_in, const int* in_sizes, int n_in,
                              void* d_out, int out_size, void* d_ws, size_t ws_size,
                              hipStream_t stream) {
  (void)in_sizes; (void)n_in; (void)out_size; (void)ws_size;

  const int*   batch_word    = (const int*)  d_in[0];
  const int*   batch_intents = (const int*)  d_in[1];
  const int*   batch_char    = (const int*)  d_in[3];
  const int*   batch_lexi    = (const int*)  d_in[6];
  const int*   batch_label   = (const int*)  d_in[7];
  const float* char_emb      = (const float*)d_in[8];
  const float* word_emb      = (const float*)d_in[9];
  const float* lexi_emb      = (const float*)d_in[10];
  const float* intent_emb    = (const float*)d_in[11];
  const float* conv_w        = (const float*)d_in[12];
  const float* conv_b        = (const float*)d_in[13];
  const float* lstm_w_ih     = (const float*)d_in[14];
  const float* lstm_w_hh     = (const float*)d_in[15];
  const float* lstm_b_ih     = (const float*)d_in[16];
  const float* lstm_b_hh     = (const float*)d_in[17];
  const float* proj_w        = (const float*)d_in[18];
  const float* proj_b        = (const float*)d_in[19];
  const float* transitions   = (const float*)d_in[20];
  float* out = (float*)d_out;

  // ---- workspace layout -------------------------------------------------
  size_t off = 0;
  auto alloc = [&](size_t bytes) {
    off = (off + 255) & ~(size_t)255;
    size_t o = off; off += bytes; return o;
  };
  char* ws = (char*)d_ws;
  const size_t o_kv16   = alloc((size_t)kBT * kD * 2);
  const size_t o_kvT16  = alloc((size_t)kB * kD * kT * 2);
  const size_t o_q16    = alloc((size_t)kBT * kD * 2);          // reused for attn_f16
  const size_t o_x0     = alloc((size_t)kBT * kLH * 2);
  const size_t o_x1     = alloc((size_t)kBT * kLH * 2);
  const size_t o_im2col = alloc((size_t)kBT * (3 * kDC) * 2);
  const size_t o_wih16  = alloc((size_t)4 * kH4 * kLH * 2);
  const size_t o_wc16   = alloc((size_t)kHC * (3 * kDC) * 2);
  const size_t o_pw16   = alloc((size_t)kGPad * kLH * 2);       // zero-padded rows
  const size_t o_nb     = alloc((size_t)4 * kH4 * 4);
  const size_t o_scores = alloc((size_t)kBT * kT * 4);
  const size_t o_pre    = alloc((size_t)kBT * kH4 * 4);         // also conv output temp
  const size_t o_emit   = alloc((size_t)kBT * kGP * 4);
  const size_t o_bp     = alloc((size_t)kB * (kT - 1) * kG * 4);
  const size_t o_lossb  = alloc((size_t)kB * 4);

  _Float16* kv16    = (_Float16*)(ws + o_kv16);
  _Float16* kvT16   = (_Float16*)(ws + o_kvT16);
  _Float16* q16     = (_Float16*)(ws + o_q16);
  _Float16* attn16  = (_Float16*)(ws + o_q16);   // reuse after q consumed
  _Float16* x0      = (_Float16*)(ws + o_x0);
  _Float16* x1      = (_Float16*)(ws + o_x1);
  _Float16* im2c    = (_Float16*)(ws + o_im2col);
  _Float16* wih16   = (_Float16*)(ws + o_wih16);
  _Float16* wc16    = (_Float16*)(ws + o_wc16);
  _Float16* pw16    = (_Float16*)(ws + o_pw16);
  float*    nb      = (float*)   (ws + o_nb);
  float*    scores  = (float*)   (ws + o_scores);
  float*    pre     = (float*)   (ws + o_pre);
  float*    convout = (float*)   (ws + o_pre);
  float*    emit    = (float*)   (ws + o_emit);
  int*      bp      = (int*)     (ws + o_bp);
  float*    lossb   = (float*)   (ws + o_lossb);

  // ---- weight conversion / bias prep ------------------------------------
  {
    const int n_wih = 4 * kH4 * kLH;
    cvt_f32_to_f16<<<512, 256, 0, stream>>>(lstm_w_ih, wih16, n_wih);
    const int n_wc = kHC * kDC * 3;   // flat (o, c, k) == GEMM row layout
    cvt_f32_to_f16<<<256, 256, 0, stream>>>(conv_w, wc16, n_wc);
    pad_proj_w<<<(kGPad * kLH + 255) / 256, 256, 0, stream>>>(proj_w, pw16);
    comb_bias<<<(4 * kH4 + 255) / 256, 256, 0, stream>>>(lstm_b_ih, lstm_b_hh, nb,
                                                         4 * kH4);
  }

  // ---- gathers ----------------------------------------------------------
  prep_kv<<<kBT, 128, 0, stream>>>(batch_word, batch_lexi, batch_intents,
                                   word_emb, lexi_emb, intent_emb, kv16, kvT16);
  im2col_char<<<kBT, 128, 0, stream>>>(batch_char, char_emb, im2c);

  // ---- char CNN as GEMM: [16384,384] x [384,384]^T + conv_b -------------
  gemm_nt_wmma<<<dim3(kHC / 64, kBT / 16, 1), 32, 0, stream>>>(
      im2c, wc16, convout, nullptr, kBT, kHC, 3 * kDC, 0, 0, 0, conv_b, kHC);

  build_q<<<kBT, 128, 0, stream>>>(convout, batch_intents, intent_emb, q16);

  // ---- attention scores: per-batch [256,512] x [256,512]^T --------------
  gemm_nt_wmma<<<dim3(kT / 64, kT / 16, kB), 32, 0, stream>>>(
      q16, kv16, scores, nullptr, kT, kT, kD,
      (long long)kT * kD, (long long)kT * kD, (long long)kT * kT, nullptr, 0);

  const float invTemp = 1.0f / 11.3137084989847604f;  // 1/sqrt(128)
  softmax_rows<<<kBT, kT, 0, stream>>>(scores, attn16, invTemp);

  // ---- context: per-batch [256,256] x [512,256]^T -> x0 (f16) -----------
  gemm_nt_wmma<<<dim3(kD / 64, kT / 16, kB), 32, 0, stream>>>(
      attn16, kvT16, nullptr, x0, kT, kD, kT,
      (long long)kT * kT, (long long)kD * kT, (long long)kT * kD, nullptr, 0);

  // ---- 2-layer bidirectional LSTM ---------------------------------------
  for (int l = 0; l < 2; ++l) {
    _Float16* xin  = (l == 0) ? x0 : x1;
    _Float16* xout = (l == 0) ? x1 : x0;
    for (int d = 0; d < 2; ++d) {
      const int ld = l * 2 + d;
      // pre = x @ w_ih^T + (b_ih + b_hh): [16384,512] x [1024,512]^T
      gemm_nt_wmma<<<dim3(kH4 / 64, kBT / 16, 1), 32, 0, stream>>>(
          xin, wih16 + (size_t)ld * kH4 * kLH, pre, nullptr,
          kBT, kH4, kLH, 0, 0, 0, nb + (size_t)ld * kH4, kH4);
      lstm_recur<<<kB, kH, 0, stream>>>(
          pre, lstm_w_hh + (size_t)ld * kH4 * kH, xout, /*reverse=*/d,
          /*colOffset=*/d * kH);
    }
  }

  // ---- output projection: [16384,512] x [128,512]^T(padded) + proj_b ----
  gemm_nt_wmma<<<dim3((kGP + 63) / 64, kBT / 16, 1), 32, 0, stream>>>(
      x0, pw16, emit, nullptr, kBT, kGP, kLH, 0, 0, 0, proj_b, kG);

  // ---- CRF NLL + Viterbi -------------------------------------------------
  crf_nll<<<kB, 128, 0, stream>>>(emit, transitions, batch_label, lossb);
  reduce_loss<<<1, kB, 0, stream>>>(lossb, out);
  viterbi<<<kB, 128, 0, stream>>>(emit, transitions, bp, out + 1);
}
